// TransitionUpdate_55173149884461
// MI455X (gfx1250) — matrix operand order repaired and checked by reading, not compiled, not run on hardware
//
#include <hip/hip_runtime.h>
#include <stdint.h>

typedef __attribute__((ext_vector_type(16))) __bf16 v16bf;
typedef __attribute__((ext_vector_type(8)))  __bf16 v8bf;
typedef __attribute__((ext_vector_type(4)))  __bf16 v4bf;
typedef __attribute__((ext_vector_type(8)))  float  v8f;

#define DIM    128
#define HIDDEN 512
#define LN_EPS 1e-5f

// One block = 256 threads = 8 wave32, one 16-token tile per iteration.
// LDS (~305 KB of CDNA5's 320 KB/WGP): bf16 weights + double-buffered async x tiles.
__global__ __launch_bounds__(256, 1)
void ln_mlp_wmma_kernel(const float* __restrict__ x,
                        const float* __restrict__ w1,   // [HIDDEN][DIM]
                        const float* __restrict__ w2,   // [DIM][HIDDEN]
                        const float* __restrict__ b1,   // [HIDDEN]
                        const float* __restrict__ b2,   // [DIM]
                        const float* __restrict__ wn,   // [DIM]
                        const float* __restrict__ bn,   // [DIM]
                        float* __restrict__ out,        // [tokens][DIM]
                        int ntiles)
{
    __shared__ __align__(16) __bf16 lds_w1[HIDDEN * DIM];   // 128 KB, row-major [h][d]
    __shared__ __align__(16) __bf16 lds_w2[DIM * HIDDEN];   // 128 KB, row-major [d][h]
    __shared__ __align__(16) float  lds_x [2][16 * DIM];    //  16 KB, async double buffer
    __shared__ __align__(16) __bf16 lds_a [16 * DIM];       //   4 KB, normalized tile
    __shared__ __align__(16) __bf16 lds_h [16 * HIDDEN];    //  16 KB, relu(h) tile
    __shared__ float lds_wn[DIM], lds_bn[DIM];
    __shared__ float lds_b1[HIDDEN], lds_b2[DIM];
    __shared__ float red_s[16][17], red_q[16][17];          // LN reduction scratch

    const int tid  = threadIdx.x;
    const int lane = tid & 31;
    const int wave = tid >> 5;        // 0..7
    const int lh   = lane >> 4;       // lane half: 0 or 1
    const int lm   = lane & 15;       // row (A) / column (B/D) index within 16

    // ---------------- Preload: weights f32 -> bf16 into LDS, LN params, biases ----------------
    for (int i = tid; i < DIM; i += 256) { lds_wn[i] = wn[i]; lds_bn[i] = bn[i]; lds_b2[i] = b2[i]; }
    for (int i = tid; i < HIDDEN; i += 256) lds_b1[i] = b1[i];
    {
        const float4* w1v = (const float4*)w1;
        for (int i = tid; i < (HIDDEN * DIM) / 4; i += 256) {
            float4 v = w1v[i];
            v4bf p = { (__bf16)v.x, (__bf16)v.y, (__bf16)v.z, (__bf16)v.w };
            *(v4bf*)&lds_w1[4 * i] = p;
        }
        const float4* w2v = (const float4*)w2;
        for (int i = tid; i < (DIM * HIDDEN) / 4; i += 256) {
            float4 v = w2v[i];
            v4bf p = { (__bf16)v.x, (__bf16)v.y, (__bf16)v.z, (__bf16)v.w };
            *(v4bf*)&lds_w2[4 * i] = p;
        }
    }

    // Async-DMA a 16x128 f32 x tile into an LDS buffer (2 x B128 per thread, ASYNCcnt +2/wave).
    // The instruction's offset is added to BOTH the LDS and global addresses.
    auto issue_x_tile = [&](long tile_idx, int buf) {
        uint32_t ldsoff = (uint32_t)(uintptr_t)&lds_x[buf][tid * 8];
        uint32_t goff   = (uint32_t)((tile_idx * 16 * DIM + tid * 8) * sizeof(float));
        asm volatile(
            "global_load_async_to_lds_b128 %0, %1, %2\n\t"
            "global_load_async_to_lds_b128 %0, %1, %2 offset:16"
            :: "v"(ldsoff), "v"(goff), "s"(x) : "memory");
    };

    // Prologue: start DMA of this block's first tile.
    if (blockIdx.x < (unsigned)ntiles) issue_x_tile(blockIdx.x, 0);

    int iter = 0;
    for (long tile = blockIdx.x; tile < ntiles; tile += gridDim.x, ++iter) {
        const long token0 = tile * 16;
        const int  cur    = iter & 1;

        // Kick off DMA of the next tile into the other buffer (dummy re-load keeps the
        // ASYNCcnt discipline exact on the final iteration; that data is never read).
        {
            long nxt = tile + gridDim.x;
            if (nxt >= ntiles) nxt = tile;
            issue_x_tile(nxt, cur ^ 1);
        }
        // <=2 outstanding left  ==>  the oldest 2 (current tile, issued last iter) are done.
        asm volatile("s_wait_asynccnt 0x2" ::: "memory");
        __syncthreads();   // make tile's LDS data visible across all 8 waves

        // ---------------- LayerNorm (from LDS) -> bf16 A tile ----------------
        {
            const int tok = tid >> 4;   // 0..15 token in tile
            const int seg = tid & 15;   // 16 threads/token, 8 elems each
            float4 u0 = *(const float4*)&lds_x[cur][tok * DIM + seg * 8];
            float4 u1 = *(const float4*)&lds_x[cur][tok * DIM + seg * 8 + 4];
            float s = u0.x + u0.y + u0.z + u0.w + u1.x + u1.y + u1.z + u1.w;
            float q = u0.x*u0.x + u0.y*u0.y + u0.z*u0.z + u0.w*u0.w
                    + u1.x*u1.x + u1.y*u1.y + u1.z*u1.z + u1.w*u1.w;
            red_s[tok][seg] = s;
            red_q[tok][seg] = q;
            __syncthreads();
            float st = 0.f, qt = 0.f;
            #pragma unroll
            for (int i = 0; i < 16; ++i) { st += red_s[tok][i]; qt += red_q[tok][i]; }
            const float mu = st * (1.0f / DIM);
            const float rs = rsqrtf(qt * (1.0f / DIM) - mu * mu + LN_EPS);
            float e[8] = { u0.x, u0.y, u0.z, u0.w, u1.x, u1.y, u1.z, u1.w };
            #pragma unroll
            for (int i = 0; i < 8; ++i) {
                const int d = seg * 8 + i;
                lds_a[tok * DIM + d] = (__bf16)((e[i] - mu) * rs * lds_wn[d] + lds_bn[d]);
            }
        }
        __syncthreads();

        // ---- A fragments (16x32 bf16, ISA layout): lane lm = row M, K-slices by lane half ----
        v16bf afr[4];
        #pragma unroll
        for (int kk = 0; kk < 4; ++kk) {
            v8bf lo = *(const v8bf*)&lds_a[lm * DIM + kk * 32 + lh * 8];        // K = 32kk+8h..+8
            v8bf hi = *(const v8bf*)&lds_a[lm * DIM + kk * 32 + 16 + lh * 8];   // K = 32kk+16+8h..+8
            #pragma unroll
            for (int i = 0; i < 8; ++i) { afr[kk][i] = lo[i]; afr[kk][i + 8] = hi[i]; }
        }

        // ---------------- GEMM1: h = relu(A @ w1^T + b1); 4 N-blocks per wave ----------------
        #pragma unroll
        for (int j = 0; j < 4; ++j) {
            const int n0   = (wave * 4 + j) * 16;   // hidden column block
            const int hidx = n0 + lm;               // B layout: N = lane%16
            v8f acc = { 0.f, 0.f, 0.f, 0.f, 0.f, 0.f, 0.f, 0.f };
            #pragma unroll
            for (int kk = 0; kk < 4; ++kk) {
                // B(k,n)=w1[n][k]: K = 32kk+8h..+8 and 32kk+16+8h..+8 at row hidx (same K order as A)
                v8bf lo = *(const v8bf*)&lds_w1[hidx * DIM + kk * 32 + lh * 8];
                v8bf hi = *(const v8bf*)&lds_w1[hidx * DIM + kk * 32 + 16 + lh * 8];
                v16bf bfr;
                #pragma unroll
                for (int i = 0; i < 8; ++i) { bfr[i] = lo[i]; bfr[i + 8] = hi[i]; }
                acc = __builtin_amdgcn_wmma_f32_16x16x32_bf16(
                          false, afr[kk], false, bfr, (short)0, acc, false, false);
            }
            const float bb = lds_b1[hidx];
            #pragma unroll
            for (int r = 0; r < 8; ++r) {          // D layout: row M = r + 8*lane_half
                float v = acc[r] + bb;
                v = v > 0.f ? v : 0.f;
                lds_h[(r + lh * 8) * HIDDEN + hidx] = (__bf16)v;
            }
        }
        __syncthreads();

        // ---------------- GEMM2: out = H @ w2^T + b2; 1 N-block per wave, K=512 ----------------
        {
            const int n0 = wave * 16;
            const int d  = n0 + lm;
            v8f acc = { 0.f, 0.f, 0.f, 0.f, 0.f, 0.f, 0.f, 0.f };
            #pragma unroll
            for (int kk = 0; kk < 16; ++kk) {
                v8bf alo = *(const v8bf*)&lds_h[lm * HIDDEN + kk * 32 + lh * 8];
                v8bf ahi = *(const v8bf*)&lds_h[lm * HIDDEN + kk * 32 + 16 + lh * 8];
                v16bf a2;
                #pragma unroll
                for (int i = 0; i < 8; ++i) { a2[i] = alo[i]; a2[i + 8] = ahi[i]; }
                // B(k,d)=w2[d][k]: K = 32kk+8h..+8 and 32kk+16+8h..+8 at row d (same K order as A)
                v8bf blo = *(const v8bf*)&lds_w2[d * HIDDEN + kk * 32 + lh * 8];
                v8bf bhi = *(const v8bf*)&lds_w2[d * HIDDEN + kk * 32 + 16 + lh * 8];
                v16bf bfr;
                #pragma unroll
                for (int i = 0; i < 8; ++i) { bfr[i] = blo[i]; bfr[i + 8] = bhi[i]; }
                acc = __builtin_amdgcn_wmma_f32_16x16x32_bf16(
                          false, a2, false, bfr, (short)0, acc, false, false);
            }
            const float bb = lds_b2[d];
            #pragma unroll
            for (int r = 0; r < 8; ++r) {
                out[(token0 + r + lh * 8) * DIM + d] = acc[r] + bb;
            }
        }
        __syncthreads();   // protect lds_a / lds_h / red_* / lds_x before next iteration
    }
}

extern "C" void kernel_launch(void* const* d_in, const int* in_sizes, int n_in,
                              void* d_out, int out_size, void* d_ws, size_t ws_size,
                              hipStream_t stream) {
    const float* x  = (const float*)d_in[0];
    const float* w1 = (const float*)d_in[1];
    const float* w2 = (const float*)d_in[2];
    const float* b1 = (const float*)d_in[3];
    const float* b2 = (const float*)d_in[4];
    const float* wn = (const float*)d_in[5];
    const float* bn = (const float*)d_in[6];
    float* out = (float*)d_out;

    const int tokens = in_sizes[0] / DIM;   // 262144
    const int ntiles = tokens / 16;         // 16384

    int grid = 1024;                        // persistent blocks; 1 block/WGP (LDS-bound)
    if (grid > ntiles) grid = ntiles;

    ln_mlp_wmma_kernel<<<grid, 256, 0, stream>>>(x, w1, w2, b1, b2, wn, bn, out, ntiles);
}
